// LatentAttentionBlock_43550968381507
// MI455X (gfx1250) — matrix operand order repaired and checked by reading, not compiled, not run on hardware
//
#include <hip/hip_runtime.h>
#include <hip/hip_bf16.h>

typedef __attribute__((ext_vector_type(16))) _Float16 v16h;
typedef __attribute__((ext_vector_type(8)))  _Float16 v8h;
typedef __attribute__((ext_vector_type(8)))  float    v8f;

// ---------------------------------------------------------------------------
// Problem constants
// ---------------------------------------------------------------------------
#define BATCH   8
#define SEQ     1024
#define DMODEL  1024
#define NHEAD   4
#define QKDIM   128          // per-head 32
#define EXPAND  2048
#define ECOLS   (2*QKDIM + 2*EXPAND)   // 4352
#define ROWS    (BATCH*SEQ)            // 8192
#define HDQK    (QKDIM/NHEAD)          // 32
#define HDV     (DMODEL/NHEAD)         // 256
#define QSCALE  0.17677669529663687f   // 1/sqrt(32)

// ---------------------------------------------------------------------------
// WMMA helpers (CDNA5 v_wmma_f32_16x16x32_f16)
// ---------------------------------------------------------------------------
__device__ __forceinline__ v8f wmma_f16(v16h a, v16h b, v8f c) {
  return __builtin_amdgcn_wmma_f32_16x16x32_f16(false, a, false, b,
                                                (short)0, c, false, false);
}

// A fragment: 16(M) x 32(K), K-major rows (row stride = ld halves).
// Per ISA: lane L: M=L&15; halves 0..7 = K[kb..kb+7], halves 8..15 = K[16+kb..],
// kb = 8*(L>=16).
__device__ __forceinline__ v16h frag_a(const _Float16* p0, int ld, int k0) {
  int lane = threadIdx.x & 31;
  int r    = lane & 15;
  int kb   = (lane >> 4) << 3;
  const _Float16* p = p0 + (size_t)r * ld + k0 + kb;
  v8h lo = *(const v8h*)(p);
  v8h hi = *(const v8h*)(p + 16);
  v16h f;
#pragma unroll
  for (int i = 0; i < 8; ++i) { f[i] = lo[i]; f[i + 8] = hi[i]; }
  return f;
}

// B fragment: 32(K) x 16(N); we hold B^T row-major (N rows, K contiguous).
// lane L: N=L&15, halves 0..7 = K[kb..kb+7], halves 8..15 = K[16+kb..], kb = 8*(L>=16).
__device__ __forceinline__ v16h frag_b(const _Float16* p0, int ld, int k0) {
  int lane = threadIdx.x & 31;
  int n    = lane & 15;
  int kb   = (lane >> 4) << 3;
  const _Float16* p = p0 + (size_t)n * ld + k0 + kb;
  v8h lo = *(const v8h*)(p);
  v8h hi = *(const v8h*)(p + 16);
  v16h f;
#pragma unroll
  for (int i = 0; i < 8; ++i) { f[i] = lo[i]; f[i + 8] = hi[i]; }
  return f;
}

// ---------------------------------------------------------------------------
// Pipelined TN GEMM core: wave computes 32(M) x 64(N), double-buffered K.
// A row-major [*,K], W row-major [N,K] (i.e. B^T), K % 64 == 0.
// ---------------------------------------------------------------------------
template <int K>
__device__ __forceinline__ void load_frags(const _Float16* A0, const _Float16* A1,
                                           const _Float16* W0, int k0,
                                           v16h a[2], v16h b[4]) {
  a[0] = frag_a(A0, K, k0);
  a[1] = frag_a(A1, K, k0);
#pragma unroll
  for (int t = 0; t < 4; ++t)
    b[t] = frag_b(W0 + (size_t)t * 16 * K, K, k0);
}

__device__ __forceinline__ void mm8(v8f acc[2][4], const v16h a[2], const v16h b[4]) {
#pragma unroll
  for (int mt = 0; mt < 2; ++mt)
#pragma unroll
    for (int nt = 0; nt < 4; ++nt)
      acc[mt][nt] = wmma_f16(a[mt], b[nt], acc[mt][nt]);
}

template <int K>
__device__ __forceinline__ void gemm_core(const _Float16* __restrict__ A,
                                          const _Float16* __restrict__ W,
                                          int m0, int n0, v8f acc[2][4]) {
  const _Float16* A0 = A + (size_t)m0 * K;
  const _Float16* A1 = A0 + (size_t)16 * K;
  const _Float16* W0 = W + (size_t)n0 * K;

  v16h aE[2], bE[4], aO[2], bO[4];
  load_frags<K>(A0, A1, W0, 0, aE, bE);
#pragma unroll 1
  for (int k0 = 0; k0 < K; k0 += 64) {
    load_frags<K>(A0, A1, W0, k0 + 32, aO, bO);   // prefetch odd chunk
    mm8(acc, aE, bE);
    if (k0 + 64 < K)
      load_frags<K>(A0, A1, W0, k0 + 64, aE, bE); // prefetch next even chunk
    mm8(acc, aO, bO);
  }
}

// C/D tile element (VGPR r, lane L): M = r + 8*(L>>4), N = L&15.
#define TILE_COORDS()                             \
  int cn  = threadIdx.x & 15;                     \
  int cmh = ((threadIdx.x >> 4) & 1) << 3;

// ---------------------------------------------------------------------------
// LayerNorm -> f16
// ---------------------------------------------------------------------------
__global__ __launch_bounds__(256) void ln_kernel(const float* __restrict__ x,
                                                 const float* __restrict__ w,
                                                 _Float16* __restrict__ xn) {
  int row = blockIdx.x;
  const float* xr = x + (size_t)row * DMODEL;
  float s = 0.f, ss = 0.f;
  for (int i = threadIdx.x; i < DMODEL; i += 256) {
    float v = xr[i]; s += v; ss += v * v;
  }
  __shared__ float rs[256], rq[256];
  rs[threadIdx.x] = s; rq[threadIdx.x] = ss;
  __syncthreads();
  for (int off = 128; off > 0; off >>= 1) {
    if (threadIdx.x < off) {
      rs[threadIdx.x] += rs[threadIdx.x + off];
      rq[threadIdx.x] += rq[threadIdx.x + off];
    }
    __syncthreads();
  }
  float mu   = rs[0] * (1.f / DMODEL);
  float var  = rq[0] * (1.f / DMODEL) - mu * mu;
  float rstd = rsqrtf(var + 1e-5f);
  for (int i = threadIdx.x; i < DMODEL; i += 256)
    xn[(size_t)row * DMODEL + i] = (_Float16)((xr[i] - mu) * rstd * w[i]);
}

// f32 -> f16 weight convert
__global__ __launch_bounds__(256) void cvt_kernel(const float* __restrict__ src,
                                                  _Float16* __restrict__ dst,
                                                  int n) {
  int i = blockIdx.x * 256 + threadIdx.x;
  if (i < n) dst[i] = (_Float16)src[i];
}

// ---------------------------------------------------------------------------
// GEMM1: h[m,n] = sum_k xn[m,k] * ew[n,k]   (M=8192, N=4352, K=1024), f16 out
// Block = 4 waves in 2x2 -> 64(M) x 128(N) block tile.
// ---------------------------------------------------------------------------
__global__ __launch_bounds__(128) void gemm_h_kernel(const _Float16* __restrict__ A,
                                                     const _Float16* __restrict__ W,
                                                     _Float16* __restrict__ Hout) {
  int wave = threadIdx.x >> 5;
  int m0   = blockIdx.y * 64  + (wave >> 1) * 32;
  int n0   = blockIdx.x * 128 + (wave & 1) * 64;
  v8f acc[2][4] = {};
  gemm_core<DMODEL>(A, W, m0, n0, acc);
  TILE_COORDS();
#pragma unroll
  for (int mt = 0; mt < 2; ++mt)
#pragma unroll
    for (int nt = 0; nt < 4; ++nt)
#pragma unroll
      for (int r = 0; r < 8; ++r)
        Hout[(size_t)(m0 + mt * 16 + cmh + r) * ECOLS + n0 + nt * 16 + cn] =
            (_Float16)acc[mt][nt][r];
}

// ---------------------------------------------------------------------------
// Pack: split h into q(scaled)/k/GEGLU; write fused[:, :1024] and V^T.
// ---------------------------------------------------------------------------
__global__ __launch_bounds__(256) void pack_kernel(const _Float16* __restrict__ h,
                                                   _Float16* __restrict__ qf,
                                                   _Float16* __restrict__ kf,
                                                   _Float16* __restrict__ vT,
                                                   _Float16* __restrict__ fused) {
  int row = blockIdx.y;                       // 0..8191
  int c   = blockIdx.x * 256 + threadIdx.x;   // 0..2303
  if (c >= 2304) return;
  int b = row >> 10, s = row & 1023;
  const _Float16* hr = h + (size_t)row * ECOLS;
  if (c < 128) {                              // q, fold 1/sqrt(32)
    int head = c >> 5, dm = c & 31;
    qf[((size_t)(b * NHEAD + head) * SEQ + s) * HDQK + dm] =
        (_Float16)((float)hr[c] * QSCALE);
  } else if (c < 256) {                       // k
    int e = c - 128; int head = e >> 5, dm = e & 31;
    kf[((size_t)(b * NHEAD + head) * SEQ + s) * HDQK + dm] = hr[c];
  } else {                                    // geglu
    int j = c - 256;                          // 0..2047
    float lin = (float)hr[256 + j];
    float pre = (float)hr[2304 + j];
    float g   = lin * (0.5f * pre * (1.f + erff(pre * 0.70710678118654752f)));
    if (j < 1024) {
      fused[(size_t)row * EXPAND + j] = (_Float16)g;           // geglu_local
    } else {
      int vj = j - 1024; int head = vj >> 8, dm = vj & 255;    // geglu_value
      // V transposed: [b,h][dim][s] so attention B-frags are contiguous
      vT[((size_t)(b * NHEAD + head) * HDV + dm) * SEQ + s] = (_Float16)g;
    }
  }
}

// ---------------------------------------------------------------------------
// Attention: one workgroup (4 waves) per (qtile, head, batch).
// Phase1: scores via WMMA + linear bias + causal -> LDS f32 [16][1024]
// Phase2: softmax (exp stored in place, 1/rowsum in LDS)
// Phase3: P @ V via WMMA (normalization folded into f16 convert)
// ---------------------------------------------------------------------------
__global__ __launch_bounds__(128) void attn_kernel(const _Float16* __restrict__ qf,
                                                   const _Float16* __restrict__ kf,
                                                   const _Float16* __restrict__ vT,
                                                   _Float16* __restrict__ fused,
                                                   const float* __restrict__ pbm) {
  __shared__ float sc[16][1024];
  __shared__ float rinv[16];

  int qt = blockIdx.x, head = blockIdx.y, b = blockIdx.z;
  int wave = threadIdx.x >> 5, lane = threadIdx.x & 31;
  size_t bh = (size_t)(b * NHEAD + head);
  const _Float16* Q  = qf + (bh * SEQ + (size_t)qt * 16) * HDQK;
  const _Float16* Kp = kf + bh * SEQ * HDQK;
  const _Float16* Vt = vT + bh * HDV * SEQ;
  float cb   = log1pf(__expf(pbm[0]));       // softplus(position_bias_mult)
  int   jmax = (qt + 1) * 16;

  // ---- phase 1: scores ----
  v16h aq = frag_a(Q, HDQK, 0);
  for (int kt = wave; kt <= qt; kt += 4) {
    v16h bk = frag_b(Kp + (size_t)kt * 16 * HDQK, HDQK, 0);
    v8f s = {};
    s = wmma_f16(aq, bk, s);
    int ih = (lane >> 4) << 3;
    int n  = lane & 15;
#pragma unroll
    for (int r = 0; r < 8; ++r) {
      int il = r + ih;
      int i  = qt * 16 + il;
      int j  = kt * 16 + n;
      float v = s[r];
      v = (j <= i) ? (v + cb * (float)(j - i)) : -1e30f;
      sc[il][kt * 16 + n] = v;
    }
  }
  __syncthreads();

  // ---- phase 2: softmax (8 lanes per row; a group lives inside one wave) ----
  int row = threadIdx.x >> 3, sub = threadIdx.x & 7;
  float mx = -1e30f;
  for (int j = sub; j < jmax; j += 8) mx = fmaxf(mx, sc[row][j]);
#pragma unroll
  for (int d = 1; d < 8; d <<= 1) mx = fmaxf(mx, __shfl_xor(mx, d, 32));
  float sm = 0.f;
  for (int j = sub; j < jmax; j += 8) {
    float e = __expf(sc[row][j] - mx);
    sc[row][j] = e;
    sm += e;
  }
#pragma unroll
  for (int d = 1; d < 8; d <<= 1) sm += __shfl_xor(sm, d, 32);
  if (sub == 0) rinv[row] = 1.f / sm;
  __syncthreads();

  // ---- phase 3: att = P @ V ; wave handles v-dims [wave*64, wave*64+64) ----
  v8f acc[4] = {};
  int M  = lane & 15;
  int kb = (lane >> 4) << 3;
  float rv = rinv[M];
  int nchunks = (jmax + 31) >> 5;
  for (int kc = 0; kc < nchunks; ++kc) {
    int j0 = kc * 32;
    v16h pa;
#pragma unroll
    for (int i = 0; i < 8; ++i) {
      int j1 = j0 + kb + i;
      int j2 = j0 + 16 + kb + i;
      pa[i]     = (j1 < jmax) ? (_Float16)(sc[M][j1] * rv) : (_Float16)0.f;
      pa[i + 8] = (j2 < jmax) ? (_Float16)(sc[M][j2] * rv) : (_Float16)0.f;
    }
#pragma unroll
    for (int t = 0; t < 4; ++t) {
      v16h bv = frag_b(Vt + (size_t)(wave * 64 + t * 16) * SEQ, SEQ, j0);
      acc[t] = wmma_f16(pa, bv, acc[t]);
    }
  }

  // ---- epilogue: fused[:, 1024 + head*256 + dim] ----
  TILE_COORDS();
#pragma unroll
  for (int t = 0; t < 4; ++t)
#pragma unroll
    for (int r = 0; r < 8; ++r) {
      size_t orow = (size_t)b * SEQ + qt * 16 + cmh + r;
      fused[orow * EXPAND + 1024 + head * HDV + wave * 64 + t * 16 + cn] =
          (_Float16)acc[t][r];
    }
}

// ---------------------------------------------------------------------------
// GEMM2 + residual: out[m,d] = x[m,d] + sum_e fused[m,e] * pw[d,e]
// (M=8192, N=1024, K=2048)
// ---------------------------------------------------------------------------
__global__ __launch_bounds__(128) void gemm_out_kernel(const _Float16* __restrict__ A,
                                                       const _Float16* __restrict__ W,
                                                       const float* __restrict__ X,
                                                       float* __restrict__ Out) {
  int wave = threadIdx.x >> 5;
  int m0   = blockIdx.y * 64  + (wave >> 1) * 32;
  int n0   = blockIdx.x * 128 + (wave & 1) * 64;
  v8f acc[2][4] = {};
  gemm_core<EXPAND>(A, W, m0, n0, acc);
  TILE_COORDS();
#pragma unroll
  for (int mt = 0; mt < 2; ++mt)
#pragma unroll
    for (int nt = 0; nt < 4; ++nt)
#pragma unroll
      for (int r = 0; r < 8; ++r) {
        size_t idx = (size_t)(m0 + mt * 16 + cmh + r) * DMODEL + n0 + nt * 16 + cn;
        Out[idx] = X[idx] + acc[mt][nt][r];
      }
}

// ---------------------------------------------------------------------------
// Launch
// ---------------------------------------------------------------------------
extern "C" void kernel_launch(void* const* d_in, const int* in_sizes, int n_in,
                              void* d_out, int out_size, void* d_ws, size_t ws_size,
                              hipStream_t stream) {
  const float* x    = (const float*)d_in[0];
  // d_in[1] attn_mask (causal) is implemented analytically
  const float* nw   = (const float*)d_in[2];
  const float* ew32 = (const float*)d_in[3];
  const float* pw32 = (const float*)d_in[4];
  const float* pbm  = (const float*)d_in[5];

  char* ws = (char*)d_ws;
  _Float16* xn    = (_Float16*)ws;  ws += (size_t)ROWS * DMODEL * 2;        // 16 MB
  _Float16* ew    = (_Float16*)ws;  ws += (size_t)ECOLS * DMODEL * 2;       //  9 MB
  _Float16* pw    = (_Float16*)ws;  ws += (size_t)DMODEL * EXPAND * 2;      //  4 MB
  _Float16* h     = (_Float16*)ws;  ws += (size_t)ROWS * ECOLS * 2;         // 71 MB
  _Float16* qf    = (_Float16*)ws;  ws += (size_t)ROWS * QKDIM * 2;         //  2 MB
  _Float16* kf    = (_Float16*)ws;  ws += (size_t)ROWS * QKDIM * 2;         //  2 MB
  _Float16* vT    = (_Float16*)ws;  ws += (size_t)ROWS * DMODEL * 2;        // 16 MB
  _Float16* fused = (_Float16*)ws;  ws += (size_t)ROWS * EXPAND * 2;        // 32 MB

  ln_kernel<<<ROWS, 256, 0, stream>>>(x, nw, xn);
  cvt_kernel<<<(ECOLS * DMODEL + 255) / 256, 256, 0, stream>>>(ew32, ew, ECOLS * DMODEL);
  cvt_kernel<<<(DMODEL * EXPAND + 255) / 256, 256, 0, stream>>>(pw32, pw, DMODEL * EXPAND);

  gemm_h_kernel<<<dim3(ECOLS / 128, ROWS / 64), 128, 0, stream>>>(xn, ew, h);

  pack_kernel<<<dim3(2304 / 256, ROWS), 256, 0, stream>>>(h, qf, kf, vT, fused);

  attn_kernel<<<dim3(SEQ / 16, NHEAD, BATCH), 128, 0, stream>>>(qf, kf, vT, fused, pbm);

  gemm_out_kernel<<<dim3(DMODEL / 128, ROWS / 64), 128, 0, stream>>>(fused, pw, x, (float*)d_out);
}
